// RNN_66460323938773
// MI455X (gfx1250) — compile-verified, matrix operand order checked by reading, not run on hardware
//
#include <hip/hip_runtime.h>

// RNN: xin = x @ Wx^T + bx  (65536x512 * 512x1024), then 512 sequential steps
// h = tanh(xin_t + h @ Wh^T), h:[128,1024].
//
// MI455X strategy:
//  - prep: Wh,Wx -> bf16 in ws, zero h0 + grid-barrier counter
//  - xproj: WMMA-bf16 GEMM writing xin (f32) directly into d_out
//  - scan: ONE persistent kernel (16 blocks co-resident) does all 512 steps.
//    Each block owns a 64-col N-slice and stages its 128 KB Wh slice into the
//    320 KB CDNA5 LDS once; steps are separated by an atomic grid barrier.
//    xin tile (read once, non-temporal) seeds the f32 accumulator; tanh
//    result overwrites the same d_out tile (non-temporal store) and a bf16
//    copy feeds the next step.

#define LATENT 1024
#define INPUT  512
#define TSTEPS 512
#define BATCH  128
#define NBLOCKS 16            // LATENT / 64
#define LDS_ROW_Q 129         // 128 uint4 per row + 1 pad (bank-conflict free)

typedef __attribute__((ext_vector_type(16))) __bf16 v16bf;
typedef __attribute__((ext_vector_type(8)))  float  v8f;

union Frag16 {
  v16bf v;
  uint4 q[2];
  unsigned u[8];
};

__device__ __forceinline__ unsigned pack2_bf16(float a, float b) {
#if __has_builtin(__builtin_amdgcn_cvt_pk_bf16_f32)
  typedef __attribute__((ext_vector_type(2))) __bf16 v2bf;
  v2bf r = __builtin_amdgcn_cvt_pk_bf16_f32(a, b);
  return __builtin_bit_cast(unsigned, r);
#else
  unsigned ua = __builtin_bit_cast(unsigned, a);
  unsigned ub = __builtin_bit_cast(unsigned, b);
  ua += 0x7FFFu + ((ua >> 16) & 1u);   // round-to-nearest-even
  ub += 0x7FFFu + ((ub >> 16) & 1u);
  return (ua >> 16) | (ub & 0xFFFF0000u);
#endif
}

__device__ __forceinline__ unsigned short to_bf16_bits(float a) {
#if __has_builtin(__builtin_amdgcn_cvt_pk_bf16_f32)
  return (unsigned short)(pack2_bf16(a, 0.0f) & 0xFFFFu);
#else
  unsigned ua = __builtin_bit_cast(unsigned, a);
  ua += 0x7FFFu + ((ua >> 16) & 1u);
  return (unsigned short)(ua >> 16);
#endif
}

// ---------------- prep: f32 -> bf16 weights, zero h0 + barrier -------------
__global__ void rnn_prep_kernel(const float* __restrict__ Wh,
                                const float* __restrict__ Wx,
                                unsigned short* __restrict__ Whb,
                                unsigned short* __restrict__ Wxb,
                                unsigned short* __restrict__ h0,
                                unsigned* __restrict__ bar) {
  const int stride = gridDim.x * blockDim.x;
  const int t0 = blockIdx.x * blockDim.x + threadIdx.x;
  // Pairwise convert (rows are even-length) to use v_cvt_pk_bf16_f32.
  for (int i = t0; i < (LATENT * LATENT) / 2; i += stride)
    ((unsigned*)Whb)[i] = pack2_bf16(Wh[2 * i], Wh[2 * i + 1]);
  for (int i = t0; i < (LATENT * INPUT) / 2; i += stride)
    ((unsigned*)Wxb)[i] = pack2_bf16(Wx[2 * i], Wx[2 * i + 1]);
  for (int i = t0; i < (BATCH * LATENT) / 2; i += stride)
    ((unsigned*)h0)[i] = 0u;
  if (t0 == 0) *bar = 0u;
}

// ---------------- xproj: xin = x @ Wx^T + bx  -> d_out (f32) ----------------
// One wave per 16(M)x64(N) tile; A (x) converted f32->bf16 on the fly,
// B (Wx bf16, 1 MB) stays hot in L2 across 4096 M-tiles.
__global__ __launch_bounds__(256) void rnn_xproj_kernel(
    const float* __restrict__ x,
    const unsigned short* __restrict__ Wxb,
    const float* __restrict__ bx,
    float* __restrict__ xout) {
  const int tid  = threadIdx.x;
  const int wave = tid >> 5;
  const int lane = tid & 31;
  const int half = lane >> 4;
  const int l16  = lane & 15;

  const int tileId = blockIdx.x * 8 + wave;   // 0 .. 65535
  const int mTile  = tileId >> 4;
  const int nTile  = tileId & 15;
  const int mBase  = mTile * 16;
  const int nBase  = nTile * 64;

  v8f c[4];
#pragma unroll
  for (int j = 0; j < 4; ++j) {
    const float bv = bx[nBase + j * 16 + l16];
#pragma unroll
    for (int i = 0; i < 8; ++i) c[j][i] = bv;
  }

  const int aRow = mBase + l16;
  for (int kB = 0; kB < INPUT; kB += 32) {
    // A fragment: rows of x, chunks at K = half*8 and 16+half*8.
    Frag16 a;
    const float* ap = x + (size_t)aRow * INPUT + kB + half * 8;
    const float4 f0 = *(const float4*)(ap);
    const float4 f1 = *(const float4*)(ap + 4);
    const float4 f2 = *(const float4*)(ap + 16);
    const float4 f3 = *(const float4*)(ap + 20);
    a.u[0] = pack2_bf16(f0.x, f0.y);
    a.u[1] = pack2_bf16(f0.z, f0.w);
    a.u[2] = pack2_bf16(f1.x, f1.y);
    a.u[3] = pack2_bf16(f1.z, f1.w);
    a.u[4] = pack2_bf16(f2.x, f2.y);
    a.u[5] = pack2_bf16(f2.z, f2.w);
    a.u[6] = pack2_bf16(f3.x, f3.y);
    a.u[7] = pack2_bf16(f3.z, f3.w);

#pragma unroll
    for (int j = 0; j < 4; ++j) {
      Frag16 b;
      const unsigned short* bp =
          Wxb + (size_t)(nBase + j * 16 + l16) * INPUT + kB + half * 8;
      b.q[0] = *(const uint4*)(bp);
      b.q[1] = *(const uint4*)(bp + 16);
      c[j] = __builtin_amdgcn_wmma_f32_16x16x32_bf16(
          false, a.v, false, b.v, (short)0, c[j], false, false);
    }
  }

  // C/D layout: c[i] -> row mBase + 8*half + i, col nBase + j*16 + l16
#pragma unroll
  for (int j = 0; j < 4; ++j) {
    const int col = nBase + j * 16 + l16;
#pragma unroll
    for (int i = 0; i < 8; ++i)
      xout[(size_t)(mBase + half * 8 + i) * LATENT + col] = c[j][i];
  }
}

// ---------------- persistent scan: all 512 steps in one kernel -------------
__global__ __launch_bounds__(256) void rnn_scan_kernel(
    float* __restrict__ out,                       // [512,128,1024] xin -> h
    const unsigned short* __restrict__ Whb,        // bf16 [1024,1024]
    unsigned short* __restrict__ hA,               // bf16 [128,1024]
    unsigned short* __restrict__ hB,               // bf16 [128,1024]
    unsigned* __restrict__ bar) {
  extern __shared__ uint4 ldsB[];  // 64 rows x LDS_ROW_Q uint4  (~129 KB)

  const int tid  = threadIdx.x;
  const int wave = tid >> 5;
  const int lane = tid & 31;
  const int half = lane >> 4;
  const int l16  = lane & 15;

  const int nBase = blockIdx.x * 64;   // this block's N slice
  const int mBase = wave * 16;         // this wave's M (batch-row) tile
  const int aRow  = mBase + l16;

  // ---- Stage this block's whole Wh slice (64 rows x 1024 K, bf16) once ----
  for (int i = tid; i < 64 * 128; i += 256) {
    const int row = i >> 7;        // 0..63
    const int k16 = i & 127;       // which 8-element (16 B) K chunk
    ldsB[row * LDS_ROW_Q + k16] =
        *(const uint4*)(Whb + (size_t)(nBase + row) * LATENT + k16 * 8);
  }
  __syncthreads();

  unsigned short* hIn  = hA;
  unsigned short* hOut = hB;

  for (int t = 0; t < TSTEPS; ++t) {
    float* xio = out + (size_t)t * BATCH * LATENT;

    // Seed accumulators with the xin tile. Read-once data: non-temporal.
    v8f c[4];
#pragma unroll
    for (int j = 0; j < 4; ++j) {
      const int col = nBase + j * 16 + l16;
#pragma unroll
      for (int i = 0; i < 8; ++i)
        c[j][i] = __builtin_nontemporal_load(
            &xio[(size_t)(mBase + half * 8 + i) * LATENT + col]);
    }

    // K loop: A from bf16 h state (L2), B fragments from LDS.
    for (int kB = 0; kB < LATENT; kB += 32) {
      Frag16 a;
      const unsigned short* ap = hIn + (size_t)aRow * LATENT + kB + half * 8;
      a.q[0] = *(const uint4*)(ap);
      a.q[1] = *(const uint4*)(ap + 16);

      const int kq = kB >> 3;  // uint4 index of this K window within a row
#pragma unroll
      for (int j = 0; j < 4; ++j) {
        Frag16 b;
        const int rowQ = (j * 16 + l16) * LDS_ROW_Q + kq + half;
        b.q[0] = ldsB[rowQ];        // K = kB + half*8 .. +7
        b.q[1] = ldsB[rowQ + 2];    // K = kB + 16 + half*8 .. +7
        c[j] = __builtin_amdgcn_wmma_f32_16x16x32_bf16(
            false, a.v, false, b.v, (short)0, c[j], false, false);
      }
    }

    // tanh; f32 h overwrites xin in d_out (write-once: non-temporal),
    // bf16 copy (re-read next step) stays regular-temporal.
#pragma unroll
    for (int j = 0; j < 4; ++j) {
      const int col = nBase + j * 16 + l16;
#pragma unroll
      for (int i = 0; i < 8; ++i) {
        const int row = mBase + half * 8 + i;
        const float v = tanhf(c[j][i]);
        __builtin_nontemporal_store(v, &xio[(size_t)row * LATENT + col]);
        hOut[(size_t)row * LATENT + col] = to_bf16_bits(v);
      }
    }

    // Prefetch next step's xin slice (one line per 64 B) to hide HBM latency
    // on the serial path.
    if (t + 1 < TSTEPS) {
      const float* nxt = out + (size_t)(t + 1) * BATCH * LATENT + nBase;
#pragma unroll
      for (int p = 0; p < 2; ++p) {
        const int line = tid * 2 + p;          // 0..511 lines in this slice
        const int row  = line >> 2;            // 0..127
        const int cho  = (line & 3) * 16;      // float offset within 64 cols
        __builtin_prefetch(nxt + (size_t)row * LATENT + cho, 0, 1);
      }
    }

    // ---- Grid barrier (monotonic counter; 16 co-resident blocks) ----
    __threadfence();
    __syncthreads();
    if (tid == 0) {
      const unsigned target = (unsigned)(t + 1) * (unsigned)NBLOCKS;
      __hip_atomic_fetch_add(bar, 1u, __ATOMIC_RELEASE,
                             __HIP_MEMORY_SCOPE_AGENT);
      while (__hip_atomic_load(bar, __ATOMIC_ACQUIRE,
                               __HIP_MEMORY_SCOPE_AGENT) < target)
        __builtin_amdgcn_s_sleep(1);
    }
    __syncthreads();
    __threadfence();

    // swap ping-pong h buffers
    unsigned short* tmp = hIn; hIn = hOut; hOut = tmp;
  }
}

// ---------------------------------------------------------------------------
extern "C" void kernel_launch(void* const* d_in, const int* in_sizes, int n_in,
                              void* d_out, int out_size, void* d_ws, size_t ws_size,
                              hipStream_t stream) {
  const float* x  = (const float*)d_in[0];   // [512,128,512]
  const float* Wx = (const float*)d_in[1];   // [1024,512]
  const float* bx = (const float*)d_in[2];   // [1024]
  const float* Wh = (const float*)d_in[3];   // [1024,1024]
  float* out = (float*)d_out;                // [512,128,1024]

  char* ws = (char*)d_ws;
  unsigned short* Whb = (unsigned short*)(ws);
  unsigned short* Wxb = (unsigned short*)(ws + (size_t)LATENT * LATENT * 2);
  unsigned short* hA  = (unsigned short*)(ws + (size_t)LATENT * LATENT * 2 +
                                          (size_t)LATENT * INPUT * 2);
  unsigned short* hB  = hA + (size_t)BATCH * LATENT;
  unsigned* bar       = (unsigned*)(hB + (size_t)BATCH * LATENT);

  rnn_prep_kernel<<<1024, 256, 0, stream>>>(Wh, Wx, Whb, Wxb, hA, bar);

  // 65536x1024 input-projection GEMM: 65536 wave-tiles of 16x64.
  rnn_xproj_kernel<<<(4096 * 16) / 8, 256, 0, stream>>>(x, Wxb, bx, out);

  // One persistent kernel for the whole 512-step scan.
  const size_t ldsBytes = (size_t)64 * LDS_ROW_Q * sizeof(uint4);  // ~129 KB
  rnn_scan_kernel<<<NBLOCKS, 256, ldsBytes, stream>>>(out, Whb, hA, hB, bar);
}
